// GPTModel_10926396801294
// MI455X (gfx1250) — compile-verified, matrix operand order checked by reading, not run on hardware
//
#include <hip/hip_runtime.h>
#include <cstdint>
#include <cstddef>

// ---------------------------------------------------------------------------
// GPT forward (nanoGPT-style) for MI455X / gfx1250, wave32 + WMMA f16->f32.
// All GEMMs run on v_wmma_f32_16x16x32_f16. Workspace ~212MB of d_ws.
// ---------------------------------------------------------------------------

#define BB   64
#define TT   256
#define CC   512
#define HH   8
#define HS   64
#define LL   8
#define VV   65
#define VPAD 128           // lm-head N padded to a multiple of 64
#define FF   2048          // 4*C
#define MM   (BB * TT)     // 16384 token rows
#define EPS  1e-5f

typedef __attribute__((ext_vector_type(16))) _Float16 v16h;
typedef __attribute__((ext_vector_type(8)))  _Float16 v8h;
typedef __attribute__((ext_vector_type(8)))  float    v8f;

// ---------------------------------------------------------------------------
// WMMA fragment loader.
// CDNA5 16-bit A-matrix 16x32 layout (ISA 7.12.2):
//   lane L: row r = L&15 ; K-halfblock kb = (L<16) ? 0 : 8
//   VGPR 0-3: K = kb..kb+7  ; VGPR 4-7: K = 16+kb..16+kb+7
// Same pattern serves the B-matrix when B is stored transposed (Wt[n][k]),
// with "row" = output column n.
// ---------------------------------------------------------------------------
__device__ __forceinline__ v16h load_frag(const _Float16* __restrict__ base,
                                          int ld, int k0) {
  const int lane = threadIdx.x & 31;
  const int r    = lane & 15;
  const int kb   = (lane < 16) ? 0 : 8;
  const _Float16* p = base + (size_t)r * ld + k0 + kb;
  const v8h lo = *(const v8h*)(p);        // K = k0+kb   .. +7  -> VGPR 0-3
  const v8h hi = *(const v8h*)(p + 16);   // K = k0+16+kb.. +7  -> VGPR 4-7
  v16h out;
#pragma unroll
  for (int i = 0; i < 8; ++i) { out[i] = lo[i]; out[8 + i] = hi[i]; }
  return out;
}

__device__ __forceinline__ v8f wmma_f16(v16h a, v16h b, v8f c) {
  return __builtin_amdgcn_wmma_f32_16x16x32_f16(
      /*neg_a=*/false, a, /*neg_b=*/false, b,
      /*c_mod=*/(short)0, c, /*reuse_a=*/false, /*reuse_b=*/false);
}

// ---------------------------------------------------------------------------
// Generic WMMA GEMM:  Y[M,N] = A[M,K] @ Wt[N,K]^T  (+bias) (+relu) (+resid)
// One wave -> 16x64 output strip (4 sub-tiles sharing the A fragment).
// N must be a multiple of 64. blockDim = 256 (8 waves).
// All j-loops have compile-time bounds so acc[] stays in static VGPRs
// (no movrel dynamic VGPR indexing around the WMMAs).
// ---------------------------------------------------------------------------
__global__ void gemm_wmma_kernel(const _Float16* __restrict__ A, int lda,
                                 const _Float16* __restrict__ Bt, int ldb,
                                 int M, int N, int K,
                                 const float* __restrict__ bias,
                                 const float* __restrict__ resid,
                                 float* __restrict__ outF32,
                                 _Float16* __restrict__ outF16,
                                 int relu) {
  const int wavesPerBlock = blockDim.x >> 5;
  const int waveId  = blockIdx.x * wavesPerBlock + (threadIdx.x >> 5);
  const int tiles_m  = M >> 4;
  const int tiles_n4 = N >> 6;                       // N % 64 == 0
  if (waveId >= tiles_m * tiles_n4) return;          // wave-uniform exit
  const int tm   = waveId % tiles_m;
  const int tn4  = waveId / tiles_m;
  const int lane = threadIdx.x & 31;

  const _Float16* Abase = A + (size_t)tm * 16 * lda;
  const _Float16* Bbase = Bt + (size_t)tn4 * 64 * ldb;

  v8f acc0 = {}, acc1 = {}, acc2 = {}, acc3 = {};

  for (int k0 = 0; k0 < K; k0 += 32) {
    const v16h a  = load_frag(Abase, lda, k0);
    const v16h b0 = load_frag(Bbase + (size_t)0  * ldb, ldb, k0);
    const v16h b1 = load_frag(Bbase + (size_t)16 * ldb, ldb, k0);
    const v16h b2 = load_frag(Bbase + (size_t)32 * ldb, ldb, k0);
    const v16h b3 = load_frag(Bbase + (size_t)48 * ldb, ldb, k0);
    acc0 = wmma_f16(a, b0, acc0);
    acc1 = wmma_f16(a, b1, acc1);
    acc2 = wmma_f16(a, b2, acc2);
    acc3 = wmma_f16(a, b3, acc3);
  }

  // D layout (ISA 7.12.2): n = lane&15 ; m = v + (lane<16 ? 0 : 8)
  const int n_local = lane & 15;
  const int m_off   = (lane < 16) ? 0 : 8;
  v8f accs[4] = {acc0, acc1, acc2, acc3};
#pragma unroll
  for (int j = 0; j < 4; ++j) {
    const int ncol = tn4 * 64 + j * 16 + n_local;
#pragma unroll
    for (int v = 0; v < 8; ++v) {
      const int row = tm * 16 + v + m_off;
      float val = accs[j][v];
      if (bias)  val += bias[ncol];
      if (relu)  val = fmaxf(val, 0.0f);
      if (resid) val += resid[(size_t)row * N + ncol];
      if (outF32) outF32[(size_t)row * N + ncol] = val;
      if (outF16) outF16[(size_t)row * N + ncol] = (_Float16)val;
    }
  }
}

// ---------------------------------------------------------------------------
// Attention: one block per (b, h, row-tile of 16). blockDim = 128 (4 waves).
//   scores = Q @ K^T * C^-0.5, causal mask, row softmax, attn = P @ V.
// Q/K fragments come straight from row-major q16/k16 (K row-major == Bt
// layout for Q@K^T). P lives in LDS; V is pre-transposed vt[b][h][d][t].
// ---------------------------------------------------------------------------
__global__ void attention_kernel(const _Float16* __restrict__ q16,
                                 const _Float16* __restrict__ k16,
                                 const _Float16* __restrict__ vt,
                                 _Float16* __restrict__ attn16) {
  const int mtile = blockIdx.x;   // 0..15
  const int h     = blockIdx.y;   // 0..7
  const int b     = blockIdx.z;   // 0..63
  const int wave  = threadIdx.x >> 5;  // 0..3
  const int lane  = threadIdx.x & 31;
  const float SCALE = 0.044194173824159216f;  // C^-0.5, faithful to reference

  __shared__ __align__(16) float    sS[16 * TT];
  __shared__ __align__(16) _Float16 sP[16 * TT];

  const _Float16* qbase =
      q16 + ((size_t)(b * TT + mtile * 16)) * CC + h * HS;

  // ---- scores: wave w covers key columns [w*64, w*64+64)
#pragma unroll
  for (int j = 0; j < 4; ++j) {
    const int ntile = wave * 4 + j;
    const _Float16* kbase =
        k16 + ((size_t)(b * TT + ntile * 16)) * CC + h * HS;
    v8f acc = {};
#pragma unroll
    for (int k0 = 0; k0 < HS; k0 += 32) {
      v16h a  = load_frag(qbase, CC, k0);
      v16h bb = load_frag(kbase, CC, k0);
      acc = wmma_f16(a, bb, acc);
    }
    const int n_local = lane & 15;
    const int m_off   = (lane < 16) ? 0 : 8;
#pragma unroll
    for (int v = 0; v < 8; ++v) {
      const int m   = v + m_off;
      const int col = ntile * 16 + n_local;
      const int row = mtile * 16 + m;
      sS[m * TT + col] = (col <= row) ? acc[v] * SCALE : -INFINITY;
    }
  }
  __syncthreads();

  // ---- row softmax (16 rows, one thread each)
  if (threadIdx.x < 16) {
    const int m = threadIdx.x;
    float mx = -INFINITY;
    for (int c = 0; c < TT; ++c) mx = fmaxf(mx, sS[m * TT + c]);
    float sum = 0.0f;
    for (int c = 0; c < TT; ++c) sum += __expf(sS[m * TT + c] - mx);
    const float inv = 1.0f / sum;
    for (int c = 0; c < TT; ++c)
      sP[m * TT + c] = (_Float16)(__expf(sS[m * TT + c] - mx) * inv);
  }
  __syncthreads();

  // ---- attn = P @ V : wave w handles head-dim sub-tile w (16 of 64)
  {
    const _Float16* vbase =
        vt + ((size_t)((b * HH + h) * HS) + wave * 16) * TT;
    v8f acc = {};
#pragma unroll
    for (int k0 = 0; k0 < TT; k0 += 32) {
      v16h a  = load_frag((const _Float16*)sP, TT, k0);
      v16h bb = load_frag(vbase, TT, k0);
      acc = wmma_f16(a, bb, acc);
    }
    const int n_local = lane & 15;
    const int m_off   = (lane < 16) ? 0 : 8;
#pragma unroll
    for (int v = 0; v < 8; ++v) {
      const int row = mtile * 16 + v + m_off;
      const int col = h * HS + wave * 16 + n_local;
      attn16[((size_t)(b * TT + row)) * CC + col] = (_Float16)acc[v];
    }
  }
}

// ---------------------------------------------------------------------------
// LayerNorm: one block (256 thr) per token row; emits f16 for WMMA A input.
// ---------------------------------------------------------------------------
__global__ void layernorm_kernel(const float* __restrict__ x,
                                 const float* __restrict__ g,
                                 const float* __restrict__ beta,
                                 _Float16* __restrict__ out) {
  const int row = blockIdx.x;
  const float* xr = x + (size_t)row * CC;
  __shared__ float red[256];
  const int tid = threadIdx.x;

  float s = 0.0f;
  for (int c = tid; c < CC; c += 256) s += xr[c];
  red[tid] = s; __syncthreads();
  for (int st = 128; st > 0; st >>= 1) {
    if (tid < st) red[tid] += red[tid + st];
    __syncthreads();
  }
  const float mean = red[0] * (1.0f / CC);
  __syncthreads();

  float vs = 0.0f;
  for (int c = tid; c < CC; c += 256) {
    const float d = xr[c] - mean;
    vs += d * d;
  }
  red[tid] = vs; __syncthreads();
  for (int st = 128; st > 0; st >>= 1) {
    if (tid < st) red[tid] += red[tid + st];
    __syncthreads();
  }
  const float rstd = rsqrtf(red[0] * (1.0f / CC) + EPS);

  for (int c = tid; c < CC; c += 256)
    out[(size_t)row * CC + c] =
        (_Float16)((xr[c] - mean) * rstd * g[c] + beta[c]);
}

// ---------------------------------------------------------------------------
// Embedding: x[b,t,:] = tok_emb[idx[b,t],:] + pos_emb[t,:]
// ---------------------------------------------------------------------------
__global__ void embed_kernel(const int* __restrict__ idx,
                             const float* __restrict__ tok,
                             const float* __restrict__ pos,
                             float* __restrict__ x) {
  const size_t i = (size_t)blockIdx.x * blockDim.x + threadIdx.x;
  if (i >= (size_t)MM * CC) return;
  const int c  = (int)(i % CC);
  const int mt = (int)(i / CC);
  const int t  = mt % TT;
  x[i] = tok[(size_t)idx[mt] * CC + c] + pos[(size_t)t * CC + c];
}

// ---------------------------------------------------------------------------
// Weight prep: W[K][N] f32 -> Wt[Npad][K] f16 (zero-pad extra columns).
// ---------------------------------------------------------------------------
__global__ void convert_wt_kernel(const float* __restrict__ W,
                                  _Float16* __restrict__ Wt,
                                  int K, int N, int Npad) {
  const size_t i = (size_t)blockIdx.x * blockDim.x + threadIdx.x;
  if (i >= (size_t)Npad * K) return;
  const int k = (int)(i % K);
  const int n = (int)(i / K);
  Wt[i] = (n < N) ? (_Float16)W[(size_t)k * N + n] : (_Float16)0.0f;
}

// ---------------------------------------------------------------------------
// V transpose: v16[B,T,C] f16 -> vt[B,H,HS,T] f16
// ---------------------------------------------------------------------------
__global__ void transpose_v_kernel(const _Float16* __restrict__ v16,
                                   _Float16* __restrict__ vt) {
  const size_t i = (size_t)blockIdx.x * blockDim.x + threadIdx.x;
  if (i >= (size_t)BB * TT * CC) return;
  const int c = (int)(i % CC);
  const int t = (int)((i / CC) % TT);
  const int b = (int)(i / ((size_t)CC * TT));
  const int h = c / HS;
  const int d = c % HS;
  vt[(((size_t)(b * HH + h)) * HS + d) * TT + t] = v16[i];
}

// ---------------------------------------------------------------------------
// Loss: one block (128 thr) per row. Writes raw logits (with bias) to d_out
// and atomically accumulates mean NLL into out_loss.
// ---------------------------------------------------------------------------
__global__ void init_loss_kernel(float* loss) { *loss = 0.0f; }

__global__ void loss_kernel(const float* __restrict__ logits_pad,
                            const float* __restrict__ lm_b,
                            const int* __restrict__ targets,
                            float* __restrict__ out_logits,
                            float* __restrict__ out_loss) {
  const int row = blockIdx.x;
  const int tid = threadIdx.x;
  __shared__ float red[128];

  float lv = 0.0f, l = -INFINITY;
  if (tid < VV) {
    lv = logits_pad[(size_t)row * VPAD + tid] + lm_b[tid];
    l  = lv;
    out_logits[(size_t)row * VV + tid] = lv;
  }
  red[tid] = l; __syncthreads();
  for (int s = 64; s > 0; s >>= 1) {
    if (tid < s) red[tid] = fmaxf(red[tid], red[tid + s]);
    __syncthreads();
  }
  const float mx = red[0]; __syncthreads();

  red[tid] = (tid < VV) ? __expf(lv - mx) : 0.0f;
  __syncthreads();
  for (int s = 64; s > 0; s >>= 1) {
    if (tid < s) red[tid] += red[tid + s];
    __syncthreads();
  }
  if (tid == 0) {
    const float lse = mx + logf(red[0]);
    const int tgt = targets[row];
    const float lt = logits_pad[(size_t)row * VPAD + tgt] + lm_b[tgt];
    atomicAdd(out_loss, (lse - lt) * (1.0f / (float)MM));
  }
}

// ---------------------------------------------------------------------------
// Host driver
// ---------------------------------------------------------------------------
extern "C" void kernel_launch(void* const* d_in, const int* in_sizes, int n_in,
                              void* d_out, int out_size, void* d_ws,
                              size_t ws_size, hipStream_t stream) {
  (void)in_sizes; (void)n_in; (void)out_size; (void)ws_size;

  const int*   idx     = (const int*)  d_in[0];
  const int*   targets = (const int*)  d_in[1];
  const float* tok_emb = (const float*)d_in[2];
  const float* pos_emb = (const float*)d_in[3];
  const float* ln1_g   = (const float*)d_in[4];
  const float* ln1_b   = (const float*)d_in[5];
  const float* wq      = (const float*)d_in[6];
  const float* wk      = (const float*)d_in[7];
  const float* wv      = (const float*)d_in[8];
  const float* wo      = (const float*)d_in[9];
  const float* bo      = (const float*)d_in[10];
  const float* ln2_g   = (const float*)d_in[11];
  const float* ln2_b   = (const float*)d_in[12];
  const float* w1      = (const float*)d_in[13];
  const float* b1      = (const float*)d_in[14];
  const float* w2      = (const float*)d_in[15];
  const float* b2      = (const float*)d_in[16];
  const float* lnf_g   = (const float*)d_in[17];
  const float* lnf_b   = (const float*)d_in[18];
  const float* lm_w    = (const float*)d_in[19];
  const float* lm_b    = (const float*)d_in[20];

  // ---- workspace carve (~212 MB total)
  char* ws = (char*)d_ws;
  size_t off = 0;
  auto carve = [&](size_t bytes) -> void* {
    void* p = ws + off;
    off = (off + bytes + 255) & ~(size_t)255;
    return p;
  };
  float*     x          = (float*)    carve((size_t)MM * CC * 4);
  _Float16*  h16        = (_Float16*) carve((size_t)MM * CC * 2);
  _Float16*  q16        = (_Float16*) carve((size_t)MM * CC * 2);
  _Float16*  k16        = (_Float16*) carve((size_t)MM * CC * 2);
  _Float16*  v16        = (_Float16*) carve((size_t)MM * CC * 2);
  _Float16*  vt         = (_Float16*) carve((size_t)MM * CC * 2);
  _Float16*  attn16     = (_Float16*) carve((size_t)MM * CC * 2);
  _Float16*  a1         = (_Float16*) carve((size_t)MM * FF * 2);
  _Float16*  wbuf       = (_Float16*) carve((size_t)FF * CC * 2);
  float*     logits_pad = (float*)    carve((size_t)MM * VPAD * 4);

  float* out_logits = (float*)d_out;
  float* out_loss   = out_logits + (size_t)MM * VV;

  auto launch_gemm = [&](const _Float16* A, int lda, const _Float16* Bt,
                         int ldb, int M, int N, int K, const float* bias,
                         const float* resid, float* o32, _Float16* o16,
                         int relu) {
    const int tiles  = (M / 16) * (N / 64);
    const int blocks = (tiles + 7) / 8;  // 8 waves / 256-thread block
    gemm_wmma_kernel<<<blocks, 256, 0, stream>>>(A, lda, Bt, ldb, M, N, K,
                                                 bias, resid, o32, o16, relu);
  };
  auto launch_conv = [&](const float* W, int K, int N, int Npad) {
    const size_t elems = (size_t)Npad * K;
    convert_wt_kernel<<<(unsigned)((elems + 255) / 256), 256, 0, stream>>>(
        W, wbuf, K, N, Npad);
  };

  // ---- embedding
  {
    const size_t elems = (size_t)MM * CC;
    embed_kernel<<<(unsigned)((elems + 255) / 256), 256, 0, stream>>>(
        idx, tok_emb, pos_emb, x);
  }

  // ---- transformer layers
  for (int l = 0; l < LL; ++l) {
    // LN1 -> h16
    layernorm_kernel<<<MM, 256, 0, stream>>>(x, ln1_g + l * CC,
                                             ln1_b + l * CC, h16);
    // Q, K, V projections (no bias)
    launch_conv(wq + (size_t)l * CC * CC, CC, CC, CC);
    launch_gemm(h16, CC, wbuf, CC, MM, CC, CC, nullptr, nullptr,
                nullptr, q16, 0);
    launch_conv(wk + (size_t)l * CC * CC, CC, CC, CC);
    launch_gemm(h16, CC, wbuf, CC, MM, CC, CC, nullptr, nullptr,
                nullptr, k16, 0);
    launch_conv(wv + (size_t)l * CC * CC, CC, CC, CC);
    launch_gemm(h16, CC, wbuf, CC, MM, CC, CC, nullptr, nullptr,
                nullptr, v16, 0);
    // V -> vt[B,H,HS,T]
    {
      const size_t elems = (size_t)MM * CC;
      transpose_v_kernel<<<(unsigned)((elems + 255) / 256), 256, 0, stream>>>(
          v16, vt);
    }
    // attention
    attention_kernel<<<dim3(TT / 16, HH, BB), 128, 0, stream>>>(q16, k16, vt,
                                                                attn16);
    // output projection + bias + residual -> x (f32, in place residual)
    launch_conv(wo + (size_t)l * CC * CC, CC, CC, CC);
    launch_gemm(attn16, CC, wbuf, CC, MM, CC, CC, bo + l * CC, x, x,
                nullptr, 0);
    // LN2 -> h16
    layernorm_kernel<<<MM, 256, 0, stream>>>(x, ln2_g + l * CC,
                                             ln2_b + l * CC, h16);
    // MLP: relu(h @ w1 + b1) -> a1 (f16)
    launch_conv(w1 + (size_t)l * CC * FF, CC, FF, FF);
    launch_gemm(h16, CC, wbuf, CC, MM, FF, CC, b1 + l * FF, nullptr,
                nullptr, a1, 1);
    // a1 @ w2 + b2 + residual -> x
    launch_conv(w2 + (size_t)l * FF * CC, FF, CC, CC);
    launch_gemm(a1, FF, wbuf, FF, MM, CC, FF, b2 + l * CC, x, x,
                nullptr, 0);
  }

  // ---- final LN + lm head
  layernorm_kernel<<<MM, 256, 0, stream>>>(x, lnf_g, lnf_b, h16);
  launch_conv(lm_w, CC, VV, VPAD);
  launch_gemm(h16, CC, wbuf, CC, MM, VPAD, CC, nullptr, nullptr,
              logits_pad, nullptr, 0);

  // ---- loss
  init_loss_kernel<<<1, 1, 0, stream>>>(out_loss);
  loss_kernel<<<MM, 128, 0, stream>>>(logits_pad, lm_b, targets, out_logits,
                                      out_loss);
}
